// HeteroMultiTaskGNN_11398843203890
// MI455X (gfx1250) — matrix operand order repaired and checked by reading, not compiled, not run on hardware
//
#include <hip/hip_runtime.h>
#include <hip/hip_bf16.h>
#include <math.h>

// ---------------------------------------------------------------------------
// HGT-style hetero GNN forward for MI455X (gfx1250, wave32, WMMA).
// Dense [N,64]x[64,64] projections run on v_wmma_f32_16x16x32_f16 with
// f32 accumulate. Weights are pre-packed to f16 in B-fragment order so the
// GEMM inner loop is pure b128 loads + wmma; activation flags are template
// parameters (no runtime selects); bias is always a valid pointer (zero
// scratch for bias-free GEMMs) so there is no branchy bias load.
// Edge softmax/aggregation is scatter-bound (atomics, L2-resident ~190MB).
// ---------------------------------------------------------------------------

typedef __attribute__((ext_vector_type(16))) _Float16 v16h;
typedef __attribute__((ext_vector_type(8)))  float    v8f;

#define NNODES 50000
#define NEDGES 500000
#define NBATCH 64
#define HIDDIM 64

__device__ __forceinline__ float gelu_tanh(float x) {
  const float c0 = 0.7978845608028654f; // sqrt(2/pi)
  float x3 = x * x * x;
  float t = tanhf(c0 * (x + 0.044715f * x3));
  return 0.5f * x * (1.0f + t);
}

__device__ __forceinline__ void atomicMaxF(float* addr, float val) {
  int* ai = reinterpret_cast<int*>(addr);
  int old = __float_as_int(*addr);
  while (__int_as_float(old) < val) {
    int prev = atomicCAS(ai, old, __float_as_int(val));
    if (prev == old) break;
    old = prev;
  }
}

// ---------------------------------------------------------------------------
// Weight packing: f32 [64,64] (or per-head [2][32][32] block-diag) ->
// f16 fragments laid out as Wh[((tile*2+ks)*32 + lane)*16 + e] so each lane
// loads its whole 16-half B fragment contiguously (2x global_load_b128).
// B-fragment element (lane,e): K = 8*(lane>>4) + (e&7) + 16*(e>>3) + 32*ks, N = 16*tile + (lane&15)
// ---------------------------------------------------------------------------
__global__ void pack_w_dense(const float* __restrict__ W, _Float16* __restrict__ Wh) {
  int t = blockIdx.x * blockDim.x + threadIdx.x;
  if (t >= 4096) return;
  int e = t & 15, lane = (t >> 4) & 31, ks = (t >> 9) & 1, tl = t >> 10;
  int K = (lane >> 4) * 8 + (e & 7) + (e >> 3) * 16 + ks * 32;
  int N = tl * 16 + (lane & 15);
  Wh[t] = (_Float16)W[K * HIDDIM + N];
}

__global__ void pack_w_blockdiag(const float* __restrict__ W, _Float16* __restrict__ Wh) {
  int t = blockIdx.x * blockDim.x + threadIdx.x;
  if (t >= 4096) return;
  int e = t & 15, lane = (t >> 4) & 31, ks = (t >> 9) & 1, tl = t >> 10;
  int K = (lane >> 4) * 8 + (e & 7) + (e >> 3) * 16 + ks * 32;
  int col = tl * 16 + (lane & 15);
  int h = col >> 5;
  int kk = K - h * 32;
  float v = (kk >= 0 && kk < 32) ? W[h * 1024 + kk * 32 + (col & 31)] : 0.0f;
  Wh[t] = (_Float16)v;
}

// ---------------------------------------------------------------------------
// C[nrows,64] = ACT_OUT( ACT_IN(A[nrows,64]) @ Wpacked + bias )
// One wave = one 16-row tile; 4 column tiles x 2 K-steps = 8 WMMA ops.
// A layout per ISA 7.12.2: lane(half,m) holds row m, halves e0..7 = K base..base+7,
// e8..15 = K base+16..base+23, base = 8*half + 32*ks -> 4x float4 loads.
// ACT_IN: 0 none, 1 gelu.  ACT_OUT: 0 none, 1 relu.  (compile-time)
// ---------------------------------------------------------------------------
template <int ACT_IN, int ACT_OUT>
__global__ __launch_bounds__(128)
void gemm_rows_n64(const float* __restrict__ A, const _Float16* __restrict__ Wh,
                   const float* __restrict__ bias, float* __restrict__ C, int nrows)
{
  const int wave = threadIdx.x >> 5;
  const int lane = threadIdx.x & 31;
  const int tile = blockIdx.x * 4 + wave;
  const int row0 = tile * 16;
  if (row0 >= nrows) return;          // wave-uniform: EXEC all-ones inside
  const int half = lane >> 4;
  const int m    = lane & 15;
  const float* Arow = A + (size_t)(row0 + m) * HIDDIM;

  v16h afrag[2];
#pragma unroll
  for (int ks = 0; ks < 2; ++ks) {
    const int base = half * 8 + ks * 32;
    float av[16];
    *(float4*)(av + 0)  = *(const float4*)(Arow + base);
    *(float4*)(av + 4)  = *(const float4*)(Arow + base + 4);
    *(float4*)(av + 8)  = *(const float4*)(Arow + base + 16);
    *(float4*)(av + 12) = *(const float4*)(Arow + base + 20);
#pragma unroll
    for (int e = 0; e < 16; ++e) {
      float x = av[e];
      if (ACT_IN == 1) x = gelu_tanh(x);
      afrag[ks][e] = (_Float16)x;
    }
  }

#pragma unroll
  for (int t = 0; t < 4; ++t) {
    const int col = t * 16 + m;
    v8f c = {};
#pragma unroll
    for (int ks = 0; ks < 2; ++ks) {
      v16h bfrag = *(const v16h*)(Wh + (((t * 2 + ks) * 32 + lane) << 4));
      c = __builtin_amdgcn_wmma_f32_16x16x32_f16(false, afrag[ks], false, bfrag,
                                                 (short)0, c, false, false);
    }
    const float bv = bias[col];       // always valid (zero scratch if no bias)
#pragma unroll
    for (int r = 0; r < 8; ++r) {                // C: M = r + 8*half, N = m
      const int row = row0 + r + 8 * half;
      float y = c[r] + bv;
      if (ACT_OUT == 1) y = fmaxf(y, 0.0f);
      C[(size_t)row * HIDDIM + col] = y;
    }
  }
}

// x:[N,1] -> out:[N,64] = x*w + b
__global__ void in_proj_kernel(const float* __restrict__ x, const float* __restrict__ w,
                               const float* __restrict__ b, float* __restrict__ out, int n) {
  int t = blockIdx.x * blockDim.x + threadIdx.x;
  if (t >= n * HIDDIM) return;
  int i = t >> 6, k = t & 63;
  out[t] = x[i] * w[k] + b[k];
}

// al[e,h] = (Q[dst] . KT[src])_h * p_rel[h]/sqrt(32); segment max into mbuf.
__global__ void edge_attn_kernel(const int* __restrict__ eidx, const float* __restrict__ Q,
                                 const float* __restrict__ KT, const float* __restrict__ prel,
                                 float* __restrict__ al, float* __restrict__ mbuf, int ne)
{
  int e = blockIdx.x * blockDim.x + threadIdx.x;
  if (e >= ne) return;
  int s = eidx[e];
  int d = eidx[ne + e];
  const float4* q4 = (const float4*)(Q + (size_t)d * HIDDIM);
  const float4* k4 = (const float4*)(KT + (size_t)s * HIDDIM);
  const float scale = 0.17677669529663687f; // 1/sqrt(32)
#pragma unroll
  for (int h = 0; h < 2; ++h) {
    float acc = 0.0f;
#pragma unroll
    for (int j = 0; j < 8; ++j) {
      float4 a = q4[h * 8 + j];
      float4 b = k4[h * 8 + j];
      acc += a.x * b.x + a.y * b.y + a.z * b.z + a.w * b.w;
    }
    acc *= prel[h] * scale;
    al[(size_t)e * 2 + h] = acc;
    atomicMaxF(&mbuf[(size_t)d * 2 + h], acc);
  }
}

__global__ void edge_exp_kernel(const int* __restrict__ eidx, const float* __restrict__ mbuf,
                                float* __restrict__ al, float* __restrict__ sbuf, int ne) {
  int e = blockIdx.x * blockDim.x + threadIdx.x;
  if (e >= ne) return;
  int d = eidx[ne + e];
#pragma unroll
  for (int h = 0; h < 2; ++h) {
    float a = __expf(al[(size_t)e * 2 + h] - mbuf[(size_t)d * 2 + h]);
    al[(size_t)e * 2 + h] = a;
    atomicAdd(&sbuf[(size_t)d * 2 + h], a);
  }
}

// agg[dst,k] += softmax_w(e,h) * VT[src,k]   (one thread per edge element)
__global__ void edge_scatter_kernel(const int* __restrict__ eidx, const float* __restrict__ al,
                                    const float* __restrict__ sbuf, const float* __restrict__ VT,
                                    float* __restrict__ agg, int ne) {
  long long t = (long long)blockIdx.x * blockDim.x + threadIdx.x;
  if (t >= (long long)ne * 64) return;
  int e = (int)(t >> 6);
  int k = (int)(t & 63);
  int h = k >> 5;
  int s = eidx[e];
  int d = eidx[ne + e];
  float w = al[(size_t)e * 2 + h] / (sbuf[(size_t)d * 2 + h] + 1e-16f);
  atomicAdd(&agg[(size_t)d * 64 + k], w * VT[(size_t)s * 64 + k]);
}

// X = layernorm(relu(sigmoid(skip)*O + (1-sigmoid(skip))*X)) * g + b   (in place)
__global__ void blend_relu_ln_kernel(const float* __restrict__ O, const float* __restrict__ skip,
                                     const float* __restrict__ g, const float* __restrict__ b,
                                     float* __restrict__ X, int n)
{
  int i = blockIdx.x * blockDim.x + threadIdx.x;
  if (i >= n) return;
  float sk = 1.0f / (1.0f + __expf(-skip[0]));
  const float4* o4 = (const float4*)(O + (size_t)i * 64);
  float4* x4 = (float4*)(X + (size_t)i * 64);
  float buf[64];
  float mu = 0.0f;
#pragma unroll
  for (int q = 0; q < 16; ++q) {
    float4 ov = o4[q];
    float4 xv = x4[q];
    float v0 = fmaxf(sk * ov.x + (1.0f - sk) * xv.x, 0.0f);
    float v1 = fmaxf(sk * ov.y + (1.0f - sk) * xv.y, 0.0f);
    float v2 = fmaxf(sk * ov.z + (1.0f - sk) * xv.z, 0.0f);
    float v3 = fmaxf(sk * ov.w + (1.0f - sk) * xv.w, 0.0f);
    buf[q * 4 + 0] = v0; buf[q * 4 + 1] = v1; buf[q * 4 + 2] = v2; buf[q * 4 + 3] = v3;
    mu += v0 + v1 + v2 + v3;
  }
  mu *= (1.0f / 64.0f);
  float var = 0.0f;
#pragma unroll
  for (int k = 0; k < 64; ++k) { float dd = buf[k] - mu; var += dd * dd; }
  var *= (1.0f / 64.0f);
  float inv = rsqrtf(var + 1e-5f);
#pragma unroll
  for (int q = 0; q < 16; ++q) {
    float4 y;
    y.x = (buf[q * 4 + 0] - mu) * inv * g[q * 4 + 0] + b[q * 4 + 0];
    y.y = (buf[q * 4 + 1] - mu) * inv * g[q * 4 + 1] + b[q * 4 + 1];
    y.z = (buf[q * 4 + 2] - mu) * inv * g[q * 4 + 2] + b[q * 4 + 2];
    y.w = (buf[q * 4 + 3] - mu) * inv * g[q * 4 + 3] + b[q * 4 + 3];
    x4[q] = y;
  }
}

// out[i] = b + X[i,:] . w    (decoder l2: 64 -> 1)
__global__ void dot64_kernel(const float* __restrict__ Xr, const float* __restrict__ w,
                             const float* __restrict__ b, float* __restrict__ out, int n) {
  int i = blockIdx.x * blockDim.x + threadIdx.x;
  if (i >= n) return;
  float acc = b[0];
  const float4* x4 = (const float4*)(Xr + (size_t)i * 64);
  const float4* w4 = (const float4*)w;
#pragma unroll
  for (int q = 0; q < 16; ++q) {
    float4 a = x4[q];
    float4 c = w4[q];
    acc += a.x * c.x + a.y * c.y + a.z * c.z + a.w * c.w;
  }
  out[i] = acc;
}

__global__ void pool_acc_kernel(const float* __restrict__ X, const int* __restrict__ batch,
                                float* __restrict__ pooled, float* __restrict__ cnt,
                                int n, int coloff, int cntoff) {
  int t = blockIdx.x * blockDim.x + threadIdx.x;
  if (t >= n * 64) return;
  int i = t >> 6, k = t & 63;
  int bi = batch[i];
  atomicAdd(&pooled[(size_t)bi * 128 + coloff + k], X[t]);
  if (k == 0) atomicAdd(&cnt[cntoff + bi], 1.0f);
}

__global__ void cls_l1_kernel(const float* __restrict__ pooled, const float* __restrict__ cnt,
                              const float* __restrict__ W, const float* __restrict__ b,
                              float* __restrict__ h1) {
  int t = blockIdx.x * blockDim.x + threadIdx.x;
  if (t >= NBATCH * 64) return;
  int bi = t >> 6, j = t & 63;
  float acc = b[j];
  for (int i = 0; i < 128; ++i) {
    float c = cnt[(i >> 6) * NBATCH + bi];
    float v = pooled[(size_t)bi * 128 + i] / fmaxf(c, 1.0f);
    acc += v * W[i * 64 + j];
  }
  h1[t] = fmaxf(acc, 0.0f);
}

__global__ void cls_l2_kernel(const float* __restrict__ h1, const float* __restrict__ W,
                              const float* __restrict__ b, float* __restrict__ logits) {
  int t = blockIdx.x * blockDim.x + threadIdx.x;
  if (t >= NBATCH * 2) return;
  int bi = t >> 1, j = t & 1;
  float acc = b[j];
  for (int k = 0; k < 64; ++k) acc += h1[bi * 64 + k] * W[k * 2 + j];
  logits[t] = acc;
}

__global__ void fill_kernel(float* __restrict__ p, float v, long long n) {
  long long t = (long long)blockIdx.x * blockDim.x + threadIdx.x;
  if (t < n) p[t] = v;
}

// ---------------------------------------------------------------------------

static inline int cdiv_i(long long a, long long b) { return (int)((a + b - 1) / b); }

extern "C" void kernel_launch(void* const* d_in, const int* in_sizes, int n_in,
                              void* d_out, int out_size, void* d_ws, size_t ws_size,
                              hipStream_t stream)
{
  (void)in_sizes; (void)n_in; (void)out_size; (void)ws_size;

  const float* x_gene  = (const float*)d_in[0];
  const float* x_prot  = (const float*)d_in[1];
  const int*   e_gg    = (const int*)d_in[2];
  const int*   e_gp    = (const int*)d_in[3];
  const int*   e_pg    = (const int*)d_in[4];
  const int*   batch_g = (const int*)d_in[5];
  const int*   batch_p = (const int*)d_in[6];

  // params: jax pytree flatten (dict keys sorted) starting at index 7.
  // top-level order: cls(4), convs(2x27), dec(8), in_proj(4), norm(4)
  const int P = 7;
  auto F = [&](int i) { return (const float*)d_in[i]; };
  const float* cls_l1_b = F(P + 0);
  const float* cls_l1_w = F(P + 1);
  const float* cls_l2_b = F(P + 2);
  const float* cls_l2_w = F(P + 3);
  const int DEC = P + 4 + 54;
  const float* dec_b1[2] = {F(DEC + 0), F(DEC + 4)};
  const float* dec_w1[2] = {F(DEC + 1), F(DEC + 5)};
  const float* dec_b2[2] = {F(DEC + 2), F(DEC + 6)};
  const float* dec_w2[2] = {F(DEC + 3), F(DEC + 7)};
  const int INP = DEC + 8;
  const float* inp_b[2] = {F(INP + 0), F(INP + 2)};
  const float* inp_w[2] = {F(INP + 1), F(INP + 3)};
  const int NRM = INP + 4;
  const float* nrm_b[2] = {F(NRM + 0), F(NRM + 2)};
  const float* nrm_g[2] = {F(NRM + 1), F(NRM + 3)};

  // workspace layout (floats); ~193 MB + 245 KB of packed f16 weights
  float* ws = (float*)d_ws;
  const size_t SB = (size_t)NNODES * HIDDIM;
  float* Xg   = ws + 0 * SB;
  float* Xp   = ws + 1 * SB;
  float* Kg   = ws + 2 * SB;
  float* Kp   = ws + 3 * SB;
  float* Qg   = ws + 4 * SB;
  float* Qp   = ws + 5 * SB;
  float* Vg   = ws + 6 * SB;
  float* Vp   = ws + 7 * SB;
  float* KTgg = ws + 8 * SB;
  float* KTgp = ws + 9 * SB;
  float* KTpg = ws + 10 * SB;
  float* VTgg = ws + 11 * SB;
  float* VTgp = ws + 12 * SB;
  float* VTpg = ws + 13 * SB;
  float* AGGg = Kg;   // K buffers dead after KT gemms -> reuse for aggregation
  float* AGGp = Kp;
  float* Og   = Vg;   // V buffers dead after VT gemms -> reuse for out-proj
  float* Op   = Vp;
  float* al_gg = ws + 14 * SB;
  float* al_gp = al_gg + 2 * (size_t)NEDGES;
  float* al_pg = al_gp + 2 * (size_t)NEDGES;
  float* m_g   = al_pg + 2 * (size_t)NEDGES;   // [N,2]; m_g..s_p contiguous
  float* m_p   = m_g + 2 * (size_t)NNODES;
  float* s_g   = m_p + 2 * (size_t)NNODES;
  float* s_p   = s_g + 2 * (size_t)NNODES;
  float* pooled = s_p + 2 * (size_t)NNODES;    // [B,128]; cnt[128] follows
  float* cnt    = pooled + NBATCH * 128;
  float* h1     = cnt + 128;                   // [B,64]
  float* zbias  = h1 + NBATCH * 64;            // [64] zeros (bias for bias-free GEMMs)
  // packed f16 weight fragments (32B-aligned: all preceding sizes are 8-float multiples)
  _Float16* wpack = (_Float16*)(zbias + 64);
  auto WP = [&](int idx) { return wpack + (size_t)idx * 4096; };
  // slots: layer L base L*14: [0..5]=k/q/v (gene,prot), [6..7]=out, [8..10]=a_rel, [11..13]=m_rel
  // slots 28,29 = dec.l1 (gene,prot)

  const int NT = NNODES;
  const int gemm_grid = cdiv_i(NT / 16, 4);                 // 16 rows/wave, 4 waves/block
  const int eg   = cdiv_i(NEDGES, 256);
  const int eg64 = cdiv_i((long long)NEDGES * 64, 256);
  const int ng64 = cdiv_i((long long)NT * 64, 256);

  // 0) zero bias scratch + pack all weight matrices to f16 fragment layout
  fill_kernel<<<1, 64, 0, stream>>>(zbias, 0.0f, 64);
  for (int L = 0; L < 2; ++L) {
    const int CB = P + 4 + 27 * L;
    const int S = L * 14;
    pack_w_dense<<<16, 256, 0, stream>>>(F(CB + 4),  WP(S + 0));   // k gene
    pack_w_dense<<<16, 256, 0, stream>>>(F(CB + 6),  WP(S + 1));   // k prot
    pack_w_dense<<<16, 256, 0, stream>>>(F(CB + 18), WP(S + 2));   // q gene
    pack_w_dense<<<16, 256, 0, stream>>>(F(CB + 20), WP(S + 3));   // q prot
    pack_w_dense<<<16, 256, 0, stream>>>(F(CB + 24), WP(S + 4));   // v gene
    pack_w_dense<<<16, 256, 0, stream>>>(F(CB + 26), WP(S + 5));   // v prot
    pack_w_dense<<<16, 256, 0, stream>>>(F(CB + 11), WP(S + 6));   // out gene
    pack_w_dense<<<16, 256, 0, stream>>>(F(CB + 13), WP(S + 7));   // out prot
    pack_w_blockdiag<<<16, 256, 0, stream>>>(F(CB + 0), WP(S + 8));   // a_rel gg
    pack_w_blockdiag<<<16, 256, 0, stream>>>(F(CB + 1), WP(S + 9));   // a_rel gp
    pack_w_blockdiag<<<16, 256, 0, stream>>>(F(CB + 2), WP(S + 10));  // a_rel pg
    pack_w_blockdiag<<<16, 256, 0, stream>>>(F(CB + 7), WP(S + 11));  // m_rel gg
    pack_w_blockdiag<<<16, 256, 0, stream>>>(F(CB + 8), WP(S + 12));  // m_rel gp
    pack_w_blockdiag<<<16, 256, 0, stream>>>(F(CB + 9), WP(S + 13));  // m_rel pg
  }
  pack_w_dense<<<16, 256, 0, stream>>>(dec_w1[0], WP(28));
  pack_w_dense<<<16, 256, 0, stream>>>(dec_w1[1], WP(29));

  // 1) input projections
  in_proj_kernel<<<ng64, 256, 0, stream>>>(x_gene, inp_w[0], inp_b[0], Xg, NT);
  in_proj_kernel<<<ng64, 256, 0, stream>>>(x_prot, inp_w[1], inp_b[1], Xp, NT);

  // 2) two HGT conv layers
  for (int L = 0; L < 2; ++L) {
    const int CB = P + 4 + 27 * L;
    const int S = L * 14;
    const float* k_b[2]  = {F(CB + 3), F(CB + 5)};
    const float* o_b[2]  = {F(CB + 10), F(CB + 12)};
    const float* prel[3] = {F(CB + 14), F(CB + 15), F(CB + 16)};
    const float* q_b[2]  = {F(CB + 17), F(CB + 19)};
    const float* skip[2] = {F(CB + 21), F(CB + 22)};
    const float* v_b[2]  = {F(CB + 23), F(CB + 25)};

    // K/Q/V projections (WMMA)
    gemm_rows_n64<0,0><<<gemm_grid, 128, 0, stream>>>(Xg, WP(S + 0), k_b[0], Kg, NT);
    gemm_rows_n64<0,0><<<gemm_grid, 128, 0, stream>>>(Xp, WP(S + 1), k_b[1], Kp, NT);
    gemm_rows_n64<0,0><<<gemm_grid, 128, 0, stream>>>(Xg, WP(S + 2), q_b[0], Qg, NT);
    gemm_rows_n64<0,0><<<gemm_grid, 128, 0, stream>>>(Xp, WP(S + 3), q_b[1], Qp, NT);
    gemm_rows_n64<0,0><<<gemm_grid, 128, 0, stream>>>(Xg, WP(S + 4), v_b[0], Vg, NT);
    gemm_rows_n64<0,0><<<gemm_grid, 128, 0, stream>>>(Xp, WP(S + 5), v_b[1], Vp, NT);

    // relation transforms: per-head block-diagonal GEMMs (WMMA, pre-packed)
    gemm_rows_n64<0,0><<<gemm_grid, 128, 0, stream>>>(Kg, WP(S + 8),  zbias, KTgg, NT);
    gemm_rows_n64<0,0><<<gemm_grid, 128, 0, stream>>>(Kg, WP(S + 9),  zbias, KTgp, NT);
    gemm_rows_n64<0,0><<<gemm_grid, 128, 0, stream>>>(Kp, WP(S + 10), zbias, KTpg, NT);
    gemm_rows_n64<0,0><<<gemm_grid, 128, 0, stream>>>(Vg, WP(S + 11), zbias, VTgg, NT);
    gemm_rows_n64<0,0><<<gemm_grid, 128, 0, stream>>>(Vg, WP(S + 12), zbias, VTgp, NT);
    gemm_rows_n64<0,0><<<gemm_grid, 128, 0, stream>>>(Vp, WP(S + 13), zbias, VTpg, NT);

    // segment buffers: m = -inf, s = 0, agg = 0 (K buffers are dead now)
    fill_kernel<<<cdiv_i(4LL * NT, 256), 256, 0, stream>>>(m_g, -INFINITY, 4LL * NT);
    fill_kernel<<<cdiv_i(4LL * NT, 256), 256, 0, stream>>>(s_g, 0.0f, 4LL * NT);
    fill_kernel<<<cdiv_i(2LL * (long long)SB, 256), 256, 0, stream>>>(AGGg, 0.0f, 2LL * (long long)SB);

    // attention logits + segment max (gene dst: gg+pg share m_g; prot dst: gp)
    edge_attn_kernel<<<eg, 256, 0, stream>>>(e_gg, Qg, KTgg, prel[0], al_gg, m_g, NEDGES);
    edge_attn_kernel<<<eg, 256, 0, stream>>>(e_gp, Qp, KTgp, prel[1], al_gp, m_p, NEDGES);
    edge_attn_kernel<<<eg, 256, 0, stream>>>(e_pg, Qg, KTpg, prel[2], al_pg, m_g, NEDGES);

    // exp + segment sum
    edge_exp_kernel<<<eg, 256, 0, stream>>>(e_gg, m_g, al_gg, s_g, NEDGES);
    edge_exp_kernel<<<eg, 256, 0, stream>>>(e_gp, m_p, al_gp, s_p, NEDGES);
    edge_exp_kernel<<<eg, 256, 0, stream>>>(e_pg, m_g, al_pg, s_g, NEDGES);

    // normalized weighted scatter of V messages
    edge_scatter_kernel<<<eg64, 256, 0, stream>>>(e_gg, al_gg, s_g, VTgg, AGGg, NEDGES);
    edge_scatter_kernel<<<eg64, 256, 0, stream>>>(e_gp, al_gp, s_p, VTgp, AGGp, NEDGES);
    edge_scatter_kernel<<<eg64, 256, 0, stream>>>(e_pg, al_pg, s_g, VTpg, AGGg, NEDGES);

    // out projection with fused gelu on input (WMMA), then skip/relu/layernorm
    gemm_rows_n64<1,0><<<gemm_grid, 128, 0, stream>>>(AGGg, WP(S + 6), o_b[0], Og, NT);
    gemm_rows_n64<1,0><<<gemm_grid, 128, 0, stream>>>(AGGp, WP(S + 7), o_b[1], Op, NT);
    blend_relu_ln_kernel<<<cdiv_i(NT, 128), 128, 0, stream>>>(Og, skip[0], nrm_g[0], nrm_b[0], Xg, NT);
    blend_relu_ln_kernel<<<cdiv_i(NT, 128), 128, 0, stream>>>(Op, skip[1], nrm_g[1], nrm_b[1], Xp, NT);
  }

  float* out = (float*)d_out;

  // 3) per-node decoders: l2(relu(l1(x)))   (l1 via WMMA with fused relu)
  gemm_rows_n64<0,1><<<gemm_grid, 128, 0, stream>>>(Xg, WP(28), dec_b1[0], Kg, NT);
  dot64_kernel<<<cdiv_i(NT, 256), 256, 0, stream>>>(Kg, dec_w2[0], dec_b2[0], out, NT);
  gemm_rows_n64<0,1><<<gemm_grid, 128, 0, stream>>>(Xp, WP(29), dec_b1[1], Kp, NT);
  dot64_kernel<<<cdiv_i(NT, 256), 256, 0, stream>>>(Kp, dec_w2[1], dec_b2[1], out + NNODES, NT);

  // 4) mean pooling + classifier
  fill_kernel<<<cdiv_i(NBATCH * 128 + 128, 256), 256, 0, stream>>>(pooled, 0.0f, NBATCH * 128 + 128);
  pool_acc_kernel<<<ng64, 256, 0, stream>>>(Xg, batch_g, pooled, cnt, NT, 0, 0);
  pool_acc_kernel<<<ng64, 256, 0, stream>>>(Xp, batch_p, pooled, cnt, NT, 64, 64);
  cls_l1_kernel<<<cdiv_i(NBATCH * 64, 256), 256, 0, stream>>>(pooled, cnt, cls_l1_w, cls_l1_b, h1);
  cls_l2_kernel<<<1, 128, 0, stream>>>(h1, cls_l2_w, cls_l2_b, out + 2 * NNODES);
}
